// StraightThroughPolicy_12575664242880
// MI455X (gfx1250) — matrix operand order repaired and checked by reading, not compiled, not run on hardware
//
#include <hip/hip_runtime.h>

typedef __attribute__((ext_vector_type(16))) _Float16 v16h;
typedef __attribute__((ext_vector_type(8)))  float    v8f;

#define HH 320
#define WW 320
#define HP 322   // padded (1-pixel zero border)
#define WP 322
#define BB 8
static constexpr size_t HWsz = (size_t)HH * WW;
static constexpr size_t PPsz = (size_t)HP * WP;

union V16H { v16h v; _Float16 h[16]; uint4 q[2]; };
union V8H  { _Float16 h[8]; uint4 q; };

__device__ __forceinline__ float sigmoidf(float x) { return 1.0f / (1.0f + __expf(-x)); }

// ---------------- conv0: 2 -> 16 channels, direct (memory bound) ----------------
__global__ __launch_bounds__(256) void conv0_direct(
    const float* __restrict__ in,   // [B,H,W,2] f32 (NHWC == reference NCHW conv)
    const float* __restrict__ w,    // [16][2][3][3]
    const float* __restrict__ bias, // [16]
    _Float16* __restrict__ outp)    // [B,H,W,16] f16 (pre-norm, unpadded)
{
    __shared__ float sw[16 * 18];
    __shared__ float sb[16];
    for (int i = threadIdx.x; i < 288; i += 256) sw[i] = w[i];
    if (threadIdx.x < 16) sb[threadIdx.x] = bias[threadIdx.x];
    __syncthreads();

    size_t idx = (size_t)blockIdx.x * 256 + threadIdx.x;
    if (idx >= (size_t)BB * HWsz) return;
    int b = (int)(idx / HWsz);
    int rem = (int)(idx % HWsz);
    int y = rem / WW, x = rem % WW;

    float acc[16];
#pragma unroll
    for (int c = 0; c < 16; c++) acc[c] = sb[c];

    for (int t = 0; t < 9; t++) {
        int yy = y + t / 3 - 1, xx = x + t % 3 - 1;
        if (yy < 0 || yy >= HH || xx < 0 || xx >= WW) continue;
        const float* src = in + (((size_t)b * HWsz) + (size_t)yy * WW + xx) * 2;
        float v0 = src[0], v1 = src[1];
#pragma unroll
        for (int c = 0; c < 16; c++)
            acc[c] += v0 * sw[c * 18 + t] + v1 * sw[c * 18 + 9 + t];
    }
    V8H lo, hi;
#pragma unroll
    for (int c = 0; c < 8; c++) { lo.h[c] = (_Float16)acc[c]; hi.h[c] = (_Float16)acc[c + 8]; }
    uint4* dst = reinterpret_cast<uint4*>(outp + idx * 16);
    dst[0] = lo.q; dst[1] = hi.q;
}

// ---------------- instance norm stats: block per (b, 8-channel group) ----------------
__global__ __launch_bounds__(256) void inorm_stats8(
    const _Float16* __restrict__ pre,  // [B,H,W,C] unpadded
    float* __restrict__ stats, int C)
{
    int cg = C / 8;
    int b  = blockIdx.x / cg;
    int c8 = blockIdx.x % cg;
    const _Float16* p = pre + ((size_t)b * HWsz) * C + (size_t)c8 * 8;

    float s[8], ss[8];
#pragma unroll
    for (int j = 0; j < 8; j++) { s[j] = 0.f; ss[j] = 0.f; }
    for (int i = threadIdx.x; i < (int)HWsz; i += 256) {
        V8H v; v.q = *reinterpret_cast<const uint4*>(p + (size_t)i * C);
#pragma unroll
        for (int j = 0; j < 8; j++) {
            float f = (float)v.h[j];
            s[j] += f; ss[j] += f * f;
        }
    }
    __shared__ float r0[256], r1[256];
    for (int j = 0; j < 8; j++) {
        r0[threadIdx.x] = s[j]; r1[threadIdx.x] = ss[j];
        __syncthreads();
        for (int st = 128; st > 0; st >>= 1) {
            if (threadIdx.x < st) { r0[threadIdx.x] += r0[threadIdx.x + st]; r1[threadIdx.x] += r1[threadIdx.x + st]; }
            __syncthreads();
        }
        if (threadIdx.x == 0) {
            int bc = b * C + c8 * 8 + j;
            float inv = 1.0f / (float)HWsz;
            float m = r0[0] * inv;
            float var = r1[0] * inv - m * m;
            stats[2 * bc]     = m;
            stats[2 * bc + 1] = rsqrtf(var + 1e-5f);
        }
        __syncthreads();
    }
}

// ---------------- zero the 1-pixel border of the padded activation buffer ----------------
__global__ __launch_bounds__(256) void zero_border(_Float16* __restrict__ act, int C)
{
    size_t idx = (size_t)blockIdx.x * 256 + threadIdx.x;
    if (idx >= (size_t)BB * PPsz) return;
    int rem = (int)(idx % PPsz);
    int y = rem / WP, x = rem % WP;
    if (y == 0 || y == HP - 1 || x == 0 || x == WP - 1) {
        const uint4 z = {0u, 0u, 0u, 0u};
        uint4* p = reinterpret_cast<uint4*>(act + idx * C);
        for (int j = 0; j < C / 8; j++) p[j] = z;
    }
}

// ---------------- instance norm apply + ReLU, write into padded interior ----------------
__global__ __launch_bounds__(256) void inorm_apply(
    const _Float16* __restrict__ pre,  // [B,H,W,C] unpadded
    const float* __restrict__ stats,
    _Float16* __restrict__ act,        // [B,HP,WP,C] padded
    int C)
{
    size_t idx = (size_t)blockIdx.x * 256 + threadIdx.x;
    size_t total = (size_t)BB * HWsz * (size_t)C / 8;
    if (idx >= total) return;
    int cg = C / 8;
    int c8 = (int)(idx % cg);
    size_t pix = idx / cg;
    int b = (int)(pix / HWsz);
    int rem = (int)(pix % HWsz);
    int y = rem / WW, x = rem % WW;

    V8H vv; vv.q = *reinterpret_cast<const uint4*>(pre + pix * C + (size_t)c8 * 8);
    V8H ov;
#pragma unroll
    for (int j = 0; j < 8; j++) {
        int c = c8 * 8 + j;
        float m = stats[2 * (b * C + c)];
        float r = stats[2 * (b * C + c) + 1];
        float v = ((float)vv.h[j] - m) * r;
        ov.h[j] = (_Float16)(v > 0.f ? v : 0.f);
    }
    *reinterpret_cast<uint4*>(
        act + (((size_t)b * HP + (y + 1)) * WP + (x + 1)) * C + (size_t)c8 * 8) = ov.q;
}

// ---------------- weight prep: (O,I,3,3) f32 -> [tap][co][ci] f16 ----------------
__global__ __launch_bounds__(256) void prep_weights(
    const float* __restrict__ w, _Float16* __restrict__ wp, int CIN, int COUT)
{
    int idx = blockIdx.x * 256 + threadIdx.x;
    int total = 9 * COUT * CIN;
    if (idx >= total) return;
    int ci = idx % CIN;
    int co = (idx / CIN) % COUT;
    int t  = idx / (CIN * COUT);
    wp[idx] = (_Float16)w[((size_t)co * CIN + ci) * 9 + t];
}

// ---------------- conv3x3: LDS-staged shifted WMMA GEMMs over zero-padded NHWC ----------------
// Block = 8 waves covering ALL COUT/16 co-tiles of an NPG*16-pixel row segment.
// The 3-row halo tile is staged once into LDS (contiguous NHWC row copies), then every
// wave reads its B fragments from LDS (ds_load_b128, immediate offsets) -> activation
// global traffic drops from ~9*(COUT/16)x to ~1.13x.
// A (16x32 f16): lane M = lane%16; K elems {0-7,16-23} (lane<16) / {8-15,24-31}.
// B (32x16 f16): lane N = lane%16; K rows {0-7,16-23} (lane<16) / {8-15,24-31}.
// CIN == 16: two taps packed per K=32 chunk (K 0-15 = tap t, K 16-31 = tap t+1).
template <int CIN, int COUT>
__global__ __launch_bounds__(256) void conv3x3_wmma(
    const _Float16* __restrict__ act,   // [B,HP,WP,CIN] padded, zero border
    const _Float16* __restrict__ wprep, // [9][COUT][CIN]
    const float* __restrict__ bias,     // [COUT]
    _Float16* __restrict__ outp)        // [B,H,W,COUT] unpadded pre-norm
{
    constexpr int NCO  = COUT / 16;     // co tiles per block
    constexpr int NPG  = 8 / NCO;       // 16-pixel groups per block
    constexpr int SPAN = NPG * 16 + 2;  // staged pixels per row (with halo)
    constexpr int ROWH = SPAN * CIN;    // halves per staged row

    __shared__ _Float16 tile[3 * ROWH];

    const int lane = threadIdx.x & 31;
    const int wave = threadIdx.x >> 5;
    const int pg   = wave % NPG;
    const int coT  = wave / NPG;
    const int p0   = blockIdx.x * (NPG * 16);   // first output pixel of block (row-aligned)
    const int y0   = p0 / WW;
    const int xblk = p0 % WW;
    const int b    = blockIdx.z;
    const int co0  = coT * 16;
    const int n  = lane & 15;   // B/D column (pixel within group)
    const int lh = lane >> 4;   // lane half selects K sub-range
    const uint4 uz = {0u, 0u, 0u, 0u};

    // ---- cooperative stage: padded rows y0..y0+2, padded cols [xblk, xblk+SPAN) ----
    {
        constexpr int rowChunks = ROWH / 8;       // uint4 chunks per row
        for (int c = threadIdx.x; c < 3 * rowChunks; c += 256) {
            int r = c / rowChunks, o = c % rowChunks;
            const uint4* src = reinterpret_cast<const uint4*>(
                act + (((size_t)b * HP + y0 + r) * WP + xblk) * CIN) + o;
            reinterpret_cast<uint4*>(&tile[r * ROWH])[o] = *src;
        }
    }
    __syncthreads();

    v8f acc = {};
    const int xl = pg * 16 + n;   // local pixel coordinate within staged tile

    if constexpr (CIN == 16) {
        const _Float16* lb = &tile[xl * 16];
        const _Float16* abase = wprep + (size_t)(co0 + n) * 16 + lh * 8;
#pragma unroll
        for (int tp = 0; tp < 5; tp++) {
            const int t0 = 2 * tp;
            const int t1 = 2 * tp + 1;
            const int off0 = ((t0 / 3) * SPAN + (t0 % 3)) * 16;
            const int off1 = (t1 < 9) ? ((t1 / 3) * SPAN + (t1 % 3)) * 16 : 0;
            V16H bf, af;
            if (t1 < 9) {
                bf.q[0] = *reinterpret_cast<const uint4*>(lb + off0 + lh * 8);
                bf.q[1] = *reinterpret_cast<const uint4*>(lb + off1 + lh * 8);
                af.q[0] = *reinterpret_cast<const uint4*>(abase + (size_t)t0 * COUT * 16);
                af.q[1] = *reinterpret_cast<const uint4*>(abase + (size_t)t1 * COUT * 16);
            } else {  // final chunk: only tap 8 in lower K half
                bf.q[0] = *reinterpret_cast<const uint4*>(lb + off0 + lh * 8);
                bf.q[1] = uz;
                af.q[0] = *reinterpret_cast<const uint4*>(abase + (size_t)t0 * COUT * 16);
                af.q[1] = uz;
            }
            acc = __builtin_amdgcn_wmma_f32_16x16x32_f16(
                false, af.v, false, bf.v, (short)0, acc, false, false);
        }
    } else {
        const _Float16* lb = &tile[xl * CIN + lh * 8];
        const _Float16* abase = wprep + (size_t)(co0 + n) * CIN + lh * 8;
#pragma unroll
        for (int t = 0; t < 9; t++) {
            const int boff = ((t / 3) * SPAN + (t % 3)) * CIN;
#pragma unroll
            for (int kc = 0; kc < CIN; kc += 32) {
                V16H bf, af;
                const uint4* bs = reinterpret_cast<const uint4*>(lb + boff + kc);
                bf.q[0] = bs[0]; bf.q[1] = bs[2];
                const _Float16* at = abase + (size_t)t * COUT * CIN + kc;
                af.q[0] = *reinterpret_cast<const uint4*>(at);
                af.q[1] = *reinterpret_cast<const uint4*>(at + 16);
                acc = __builtin_amdgcn_wmma_f32_16x16x32_f16(
                    false, af.v, false, bf.v, (short)0, acc, false, false);
            }
        }
    }

    // D layout: VGPR v -> co = co0 + 8*lh + v ; N = lane%16
    const int coBase = co0 + lh * 8;
    V8H ov;
#pragma unroll
    for (int v = 0; v < 8; v++) ov.h[v] = (_Float16)(acc[v] + bias[coBase + v]);
    *reinterpret_cast<uint4*>(
        outp + (((size_t)b * HWsz) + (size_t)y0 * WW + xblk + xl) * COUT + coBase) = ov.q;
}

// ---------------- 32x32 avg pool (10x10 windows) over padded act, write [b][c][ph][pw] ----------------
__global__ __launch_bounds__(256) void avgpool(
    const _Float16* __restrict__ act,  // [B,HP,WP,128] padded
    float* __restrict__ pooled)        // [B,128,32,32]
{
    size_t idx = (size_t)blockIdx.x * 256 + threadIdx.x;
    if (idx >= (size_t)BB * 32 * 32 * 128) return;
    int c  = (int)(idx & 127);
    int pw = (int)((idx >> 7) & 31);
    int ph = (int)((idx >> 12) & 31);
    int b  = (int)(idx >> 17);
    float s = 0.f;
    for (int i = 0; i < 10; i++) {
        const _Float16* row =
            act + (((size_t)b * HP + (ph * 10 + i + 1)) * WP + (pw * 10 + 1)) * 128 + c;
#pragma unroll
        for (int j = 0; j < 10; j++) s += (float)row[(size_t)j * 128];
    }
    pooled[(((size_t)b * 128 + c) * 32 + ph) * 32 + pw] = s * 0.01f;
}

// ---------------- fc layer: out[b,o] = dot(x[b,:], wt[o,:]) + bias[o]; block per o ----------------
__global__ __launch_bounds__(256) void fc_layer(
    const float* __restrict__ x,  // [8][K]
    const float* __restrict__ wt, // [N][K]
    const float* __restrict__ bias,
    float* __restrict__ out,      // [8][N]
    int N, int K)
{
    int o = blockIdx.x;
    const float* wr = wt + (size_t)o * K;
    float acc[BB];
#pragma unroll
    for (int b = 0; b < BB; b++) acc[b] = 0.f;
    for (int i = threadIdx.x * 4; i < K; i += 1024) {
        __builtin_prefetch(wr + i + 4096, 0, 1);   // stream-ahead on the 512MB weight read
        float4 w4 = *reinterpret_cast<const float4*>(wr + i);
#pragma unroll
        for (int b = 0; b < BB; b++) {
            float4 x4 = *reinterpret_cast<const float4*>(x + (size_t)b * K + i);
            acc[b] += w4.x * x4.x + w4.y * x4.y + w4.z * x4.z + w4.w * x4.w;
        }
    }
    __shared__ float red[256];
#pragma unroll
    for (int b = 0; b < BB; b++) {
        red[threadIdx.x] = acc[b];
        __syncthreads();
        for (int st = 128; st > 0; st >>= 1) {
            if (threadIdx.x < st) red[threadIdx.x] += red[threadIdx.x + st];
            __syncthreads();
        }
        if (threadIdx.x == 0) out[(size_t)b * N + o] = red[0] + bias[o];
        __syncthreads();
    }
}

// ---------------- fc2 + policy head (block per batch, 320 threads) ----------------
__global__ __launch_bounds__(320) void fc2_policy(
    const float* __restrict__ x,    // [8][1024]
    const float* __restrict__ wt,   // [320][1024]
    const float* __restrict__ bias, // [320]
    const float* __restrict__ mask, // [8,1,320,320,1]
    const float* __restrict__ u,    // [8][320]
    float* __restrict__ dout)       // [binm(8,320) | prob(8,320)]
{
    int b = blockIdx.x;
    int o = threadIdx.x;
    const float* xr = x + (size_t)b * 1024;
    const float* wr = wt + (size_t)o * 1024;
    float acc = bias[o];
    for (int i = 0; i < 1024; i += 4) {
        float4 w4 = *reinterpret_cast<const float4*>(wr + i);
        float4 x4 = *reinterpret_cast<const float4*>(xr + i);
        acc += w4.x * x4.x + w4.y * x4.y + w4.z * x4.z + w4.w * x4.w;
    }
    float am = mask[(size_t)b * HWsz + o];          // mask[:,0,0,:,0]
    float prob = sigmoidf(10.f * acc) * (1.f - am);
    float una = (am == 0.f) ? 1.f : 0.f;

    __shared__ float s1[512], s2[512];
    s1[o] = una; s2[o] = prob * una;
    if (o < 192) { s1[320 + o] = 0.f; s2[320 + o] = 0.f; }
    __syncthreads();
    for (int st = 256; st > 0; st >>= 1) {
        if (o < st) { s1[o] += s1[o + st]; s2[o] += s2[o + st]; }
        __syncthreads();
    }
    float cnt  = s1[0];
    float xbar = s2[0] / cnt;
    float sp   = 32.f / cnt;
    float r    = sp / xbar;
    float beta = (1.f - sp) / (1.f - xbar);
    float resc = (r <= 1.f) ? prob * r : 1.f - (1.f - prob) * beta;
    prob = (una > 0.f) ? resc : prob;
    float uu = u[(size_t)b * WW + o];
    float bin = (prob > uu) ? 1.f : 0.f;   // binm == s + stopgrad(bin - s) == bin numerically
    dout[(size_t)b * 320 + o] = bin;
    dout[2560 + (size_t)b * 320 + o] = prob;
}

extern "C" void kernel_launch(void* const* d_in, const int* in_sizes, int n_in,
                              void* d_out, int out_size, void* d_ws, size_t ws_size,
                              hipStream_t stream)
{
    (void)in_sizes; (void)n_in; (void)out_size; (void)ws_size;
    const float* kspace = (const float*)d_in[0];
    const float* mask   = (const float*)d_in[1];
    const float* u      = (const float*)d_in[2];
    const float* c0w = (const float*)d_in[3];  const float* c0b = (const float*)d_in[4];
    const float* c1w = (const float*)d_in[5];  const float* c1b = (const float*)d_in[6];
    const float* c2w = (const float*)d_in[7];  const float* c2b = (const float*)d_in[8];
    const float* c3w = (const float*)d_in[9];  const float* c3b = (const float*)d_in[10];
    const float* f0w = (const float*)d_in[11]; const float* f0b = (const float*)d_in[12];
    const float* f1w = (const float*)d_in[13]; const float* f1b = (const float*)d_in[14];
    const float* f2w = (const float*)d_in[15]; const float* f2b = (const float*)d_in[16];
    float* out = (float*)d_out;

    char* wsp = (char*)d_ws;
    size_t off = 0;
    auto walloc = [&](size_t bytes) -> char* {
        char* p = wsp + off;
        off += (bytes + 255) & ~(size_t)255;
        return p;
    };
    const size_t preMax = (size_t)BB * HWsz * 128 * sizeof(_Float16);   // unpadded pre-norm
    const size_t actMax = (size_t)BB * PPsz * 128 * sizeof(_Float16);   // padded activations
    _Float16* bufPre = (_Float16*)walloc(preMax);
    _Float16* bufAct = (_Float16*)walloc(actMax);
    _Float16* wprep  = (_Float16*)walloc((size_t)9 * 128 * 64 * sizeof(_Float16));
    float* stats  = (float*)walloc((size_t)BB * 128 * 2 * sizeof(float));
    float* pooled = (float*)walloc((size_t)BB * 131072 * sizeof(float));
    float* fc0o   = (float*)walloc((size_t)BB * 1024 * sizeof(float));
    float* fc1o   = (float*)walloc((size_t)BB * 1024 * sizeof(float));

    const unsigned pixBlocks = (unsigned)((BB * HWsz + 255) / 256);
    const unsigned padBlocks = (unsigned)((BB * PPsz + 255) / 256);
    auto applyBlocks = [](int C) -> unsigned {
        return (unsigned)(((size_t)BB * HWsz * C / 8 + 255) / 256);
    };

    // layer 0: direct conv (2 -> 16)
    conv0_direct<<<pixBlocks, 256, 0, stream>>>(kspace, c0w, c0b, bufPre);
    inorm_stats8<<<BB * 16 / 8, 256, 0, stream>>>(bufPre, stats, 16);
    zero_border<<<padBlocks, 256, 0, stream>>>(bufAct, 16);
    inorm_apply<<<applyBlocks(16), 256, 0, stream>>>(bufPre, stats, bufAct, 16);

    // layer 1: 16 -> 32 (WMMA, paired taps; 4 pixel groups x 2 co-tiles per block)
    prep_weights<<<(9 * 32 * 16 + 255) / 256, 256, 0, stream>>>(c1w, wprep, 16, 32);
    conv3x3_wmma<16, 32><<<dim3((unsigned)(HWsz / 64), 1, BB), 256, 0, stream>>>(bufAct, wprep, c1b, bufPre);
    inorm_stats8<<<BB * 32 / 8, 256, 0, stream>>>(bufPre, stats, 32);
    zero_border<<<padBlocks, 256, 0, stream>>>(bufAct, 32);
    inorm_apply<<<applyBlocks(32), 256, 0, stream>>>(bufPre, stats, bufAct, 32);

    // layer 2: 32 -> 64 (WMMA; 2 pixel groups x 4 co-tiles per block)
    prep_weights<<<(9 * 64 * 32 + 255) / 256, 256, 0, stream>>>(c2w, wprep, 32, 64);
    conv3x3_wmma<32, 64><<<dim3((unsigned)(HWsz / 32), 1, BB), 256, 0, stream>>>(bufAct, wprep, c2b, bufPre);
    inorm_stats8<<<BB * 64 / 8, 256, 0, stream>>>(bufPre, stats, 64);
    zero_border<<<padBlocks, 256, 0, stream>>>(bufAct, 64);
    inorm_apply<<<applyBlocks(64), 256, 0, stream>>>(bufPre, stats, bufAct, 64);

    // layer 3: 64 -> 128 (WMMA; 1 pixel group x 8 co-tiles per block)
    prep_weights<<<(9 * 128 * 64 + 255) / 256, 256, 0, stream>>>(c3w, wprep, 64, 128);
    conv3x3_wmma<64, 128><<<dim3((unsigned)(HWsz / 16), 1, BB), 256, 0, stream>>>(bufAct, wprep, c3b, bufPre);
    inorm_stats8<<<BB * 128 / 8, 256, 0, stream>>>(bufPre, stats, 128);
    zero_border<<<padBlocks, 256, 0, stream>>>(bufAct, 128);
    inorm_apply<<<applyBlocks(128), 256, 0, stream>>>(bufPre, stats, bufAct, 128);

    // pool + MLP + policy
    avgpool<<<(unsigned)((BB * 32 * 32 * 128 + 255) / 256), 256, 0, stream>>>(bufAct, pooled);
    fc_layer<<<1024, 256, 0, stream>>>(pooled, f0w, f0b, fc0o, 1024, 131072);
    fc_layer<<<1024, 256, 0, stream>>>(fc0o, f1w, f1b, fc1o, 1024, 1024);
    fc2_policy<<<BB, 320, 0, stream>>>(fc1o, f2w, f2b, mask, u, out);
}
